// RoiTarget_335007450058
// MI455X (gfx1250) — matrix operand order repaired and checked by reading, not compiled, not run on hardware
//
#include <hip/hip_runtime.h>
#include <cstddef>

#define B_ 8
#define P_ 100
#define N_ 1000
#define C_ 81
#define H_ 128
#define S_ 512
#define M_ 28
#define MAXPOS_ 128

typedef __attribute__((ext_vector_type(16))) _Float16 v16h;
typedef __attribute__((ext_vector_type(8)))  float    v8f;

// Output layout (floats), concatenated in return order:
// state (8,512,1) | y (8,512,81) | bb (8,512,4) | mask (8,512,28,28,1) | prop (8,512,4)
static constexpr size_t OFF_STATE = 0;
static constexpr size_t OFF_Y     = 4096;
static constexpr size_t OFF_BB    = 4096 + 331776;            // 335872
static constexpr size_t OFF_MASK  = 335872 + 16384;           // 352256
static constexpr size_t OFF_PROP  = 352256 + 3211264;         // 3563520

// ---------------------------------------------------------------------------
// Kernel 1: IoU max/argmax per proposal. grid (ceil(N/256), B), block 256.
// ---------------------------------------------------------------------------
__global__ void k_iou(const float* __restrict__ prop, const float* __restrict__ bb,
                      float* __restrict__ mx, int* __restrict__ arg) {
  __shared__ float sB[P_ * 4];
  __shared__ float sArea[P_];
  __shared__ int   sVal[P_];
  int b = blockIdx.y;
  int t = threadIdx.x;
  if (t < P_) {
    float x1 = bb[(b * P_ + t) * 4 + 0], y1 = bb[(b * P_ + t) * 4 + 1];
    float x2 = bb[(b * P_ + t) * 4 + 2], y2 = bb[(b * P_ + t) * 4 + 3];
    sB[t * 4 + 0] = x1; sB[t * 4 + 1] = y1; sB[t * 4 + 2] = x2; sB[t * 4 + 3] = y2;
    sArea[t] = (x2 - x1) * (y2 - y1);
    sVal[t] = (x1 > 0.f) || (y1 > 0.f) || (x2 > 0.f) || (y2 > 0.f);
  }
  __syncthreads();
  int j = blockIdx.x * blockDim.x + t;
  if (j >= N_) return;
  float px1 = prop[(b * N_ + j) * 4 + 0], py1 = prop[(b * N_ + j) * 4 + 1];
  float px2 = prop[(b * N_ + j) * 4 + 2], py2 = prop[(b * N_ + j) * 4 + 3];
  float parea = (px2 - px1) * (py2 - py1);
  bool  vp = (px1 > 0.f) || (py1 > 0.f) || (px2 > 0.f) || (py2 > 0.f);
  float best = -2.0f; int barg = 0;
  for (int p = 0; p < P_; ++p) {
    float v;
    if (sVal[p]) {
      float ix1 = fmaxf(px1, sB[p * 4 + 0]);
      float iy1 = fmaxf(py1, sB[p * 4 + 1]);
      float ix2 = fminf(px2, sB[p * 4 + 2]);
      float iy2 = fminf(py2, sB[p * 4 + 3]);
      float inter = fmaxf(ix2 - ix1, 0.f) * fmaxf(iy2 - iy1, 0.f);
      float uni = parea + sArea[p] - inter;
      v = inter / fmaxf(uni, 1e-12f);
    } else {
      v = -1.0f;
    }
    if (v > best) { best = v; barg = p; }  // first max wins (strict >)
  }
  mx[b * N_ + j] = vp ? best : -1.0f;
  arg[b * N_ + j] = barg;
}

// ---------------------------------------------------------------------------
// Kernel 2: sampling / stable partition. grid B, block 256.
// ---------------------------------------------------------------------------
__global__ void k_sample(const float* __restrict__ bb, const float* __restrict__ mx,
                         const int* __restrict__ arg,
                         int* __restrict__ slot_i, int* __restrict__ slot_ti,
                         int* __restrict__ slot_fl) {
  __shared__ unsigned char sPos[N_ + P_];
  __shared__ unsigned char sNeg[N_ + P_];
  int b = blockIdx.x;
  for (int i = threadIdx.x; i < N_ + P_; i += blockDim.x) {
    bool pos = false, neg = false;
    if (i < N_) {
      float m = mx[b * N_ + i];
      pos = (m >= 0.5f);
      neg = (m < 0.5f) && (m >= 0.0f);
    } else {
      int p = i - N_;
      float x1 = bb[(b * P_ + p) * 4 + 0], y1 = bb[(b * P_ + p) * 4 + 1];
      float x2 = bb[(b * P_ + p) * 4 + 2], y2 = bb[(b * P_ + p) * 4 + 3];
      pos = (x1 > 0.f) || (y1 > 0.f) || (x2 > 0.f) || (y2 > 0.f);
    }
    sPos[i] = pos ? 1 : 0;
    sNeg[i] = neg ? 1 : 0;
  }
  __syncthreads();
  if (threadIdx.x != 0) return;
  int posTot = 0, negTot = 0;
  for (int i = 0; i < N_ + P_; ++i) { posTot += sPos[i]; negTot += sNeg[i]; }
  int n_pos  = posTot < MAXPOS_ ? posTot : MAXPOS_;
  int negCap = S_ - n_pos;
  int C0 = n_pos;
  int C1 = negTot < negCap ? negTot : negCap;
  int posSeen = 0, negSeen = 0, g0 = 0, g1 = 0, g2 = 0;
  for (int i = 0; i < N_ + P_; ++i) {
    bool sp = false, sn = false;
    if (sPos[i]) { sp = posSeen < MAXPOS_; posSeen++; }
    if (sNeg[i]) { sn = negSeen < negCap;  negSeen++; }
    int slot;
    if (sp)      slot = g0++;
    else if (sn) slot = C0 + g1++;
    else         slot = C0 + C1 + g2++;
    if (slot < S_) {
      int o = b * S_ + slot;
      slot_i[o]  = i;
      slot_ti[o] = (i < N_) ? arg[b * N_ + i] : (i - N_);
      slot_fl[o] = sp ? 1 : (sn ? 2 : 0);
    }
  }
}

// ---------------------------------------------------------------------------
// Kernel 3: state / labels / boxes / proposals. grid B*S, block 128.
// ---------------------------------------------------------------------------
__global__ void k_scalar(const float* __restrict__ y_true, const float* __restrict__ bb,
                         const float* __restrict__ props,
                         const int* __restrict__ slot_i, const int* __restrict__ slot_ti,
                         const int* __restrict__ slot_fl, float* __restrict__ out) {
  int bs = blockIdx.x;
  int b  = bs >> 9;
  int t  = threadIdx.x;
  int i  = slot_i[bs];
  int ti = slot_ti[bs];
  int fl = slot_fl[bs];
  bool sp = (fl == 1), sn = (fl == 2);
  if (t == 0) out[OFF_STATE + bs] = sp ? 1.0f : (sn ? -1.0f : 0.0f);
  if (t < C_) {
    float v = sp ? y_true[((size_t)(b * P_ + ti)) * C_ + t] : (t == 0 ? 1.0f : 0.0f);
    out[OFF_Y + (size_t)bs * C_ + t] = v;
  } else if (t < C_ + 4) {
    int c = t - C_;
    out[OFF_BB + (size_t)bs * 4 + c] = sp ? bb[(b * P_ + ti) * 4 + c] : 0.0f;
  } else if (t < C_ + 8) {
    int c = t - C_ - 4;
    float v = 0.0f;
    if (sp || sn)
      v = (i < N_) ? props[(b * N_ + i) * 4 + c] : bb[(b * P_ + (i - N_)) * 4 + c];
    out[OFF_PROP + (size_t)bs * 4 + c] = v;
  }
}

// ---------------------------------------------------------------------------
// Kernel 4: mask crop-resize as WMMA GEMM. grid B*S, block 128 (4 waves).
// R = y-lerp(mask rows)  (exact f32, -> f16)   : 32x128
// Wx = x-interp weights                        : 32x128 (row c = weights over x)
// Out(32x32) = R x Wx^T via v_wmma_f32_16x16x32_f16, one 16x16 tile per wave.
// ---------------------------------------------------------------------------
__global__ void k_mask(const float* __restrict__ mask,
                       const float* __restrict__ bb,
                       const float* __restrict__ props,
                       const int* __restrict__ slot_i,
                       const int* __restrict__ slot_ti,
                       const int* __restrict__ slot_fl,
                       float* __restrict__ out) {
  __shared__ _Float16 sR[32 * 128];
  __shared__ _Float16 sW[32 * 128];
  __shared__ float sLy[M_], sLx[M_];
  __shared__ int sY0[M_], sY1[M_], sX0[M_], sX1[M_];
  __shared__ float sBox[4];
  __shared__ int sTi, sFlag;

  int bs = blockIdx.x;
  int b  = bs >> 9;
  int t  = threadIdx.x;

  if (t == 0) {
    int i  = slot_i[bs];
    sTi    = slot_ti[bs];
    sFlag  = slot_fl[bs];
    const float* pr = (i < N_) ? &props[(b * N_ + i) * 4] : &bb[(b * P_ + (i - N_)) * 4];
    sBox[0] = pr[0]; sBox[1] = pr[1]; sBox[2] = pr[2]; sBox[3] = pr[3];
  }
  __syncthreads();

  float* outm = out + OFF_MASK + (size_t)bs * (M_ * M_);
  if (sFlag != 1) {               // not a selected positive: zero mask
    for (int idx = t; idx < M_ * M_; idx += 128) outm[idx] = 0.0f;
    return;
  }

  if (t < M_) {
    float rr = (float)t / 27.0f;
    float ys = (sBox[1] + (sBox[3] - sBox[1]) * rr) * 127.0f;
    float y0 = floorf(ys);
    sLy[t] = ys - y0;
    int y0i = (int)y0;
    sY0[t] = min(max(y0i, 0), 127);
    sY1[t] = min(max(y0i + 1, 0), 127);
    float xs = (sBox[0] + (sBox[2] - sBox[0]) * rr) * 127.0f;
    float x0 = floorf(xs);
    sLx[t] = xs - x0;
    int x0i = (int)x0;
    sX0[t] = min(max(x0i, 0), 127);
    sX1[t] = min(max(x0i + 1, 0), 127);
  }
  for (int idx = t; idx < 32 * 128; idx += 128) sW[idx] = (_Float16)0.0f;
  __syncthreads();

  // Build R: row r, col x = t (fully coalesced 512B row loads; mask set fits in L2)
  int ti = sTi;
  const float* mbase = mask + ((size_t)(b * P_ + ti)) * (H_ * H_);
  for (int r = 0; r < M_; ++r) {
    float w  = sLy[r];
    float v0 = mbase[sY0[r] * H_ + t];
    float v1 = mbase[sY1[r] * H_ + t];
    sR[r * 128 + t] = (_Float16)((1.0f - w) * v0 + w * v1);
  }
  for (int r = M_; r < 32; ++r) sR[r * 128 + t] = (_Float16)0.0f;

  // Scatter x-interp weights into Wx row c (one owner thread per row -> no races)
  if (t < M_) {
    float w = sLx[t];
    int xa = sX0[t], xb = sX1[t];
    if (xa == xb) sW[t * 128 + xa] = (_Float16)1.0f;       // (1-w)+w at clipped border
    else { sW[t * 128 + xa] = (_Float16)(1.0f - w); sW[t * 128 + xb] = (_Float16)w; }
  }
  __syncthreads();

  // WMMA: wave wv computes output tile (tM, tN); K = 128 -> 4 k-steps of 32
  const unsigned int* ru = (const unsigned int*)sR;
  const unsigned int* wu = (const unsigned int*)sW;
  int lane = t & 31;
  int wv   = t >> 5;
  int tM = wv >> 1, tN = wv & 1;
  int hf  = lane >> 4;           // half-wave selects K sub-range per ISA layout
  int l15 = lane & 15;
  int arow = tM * 16 + l15;      // A: M = lane&15 for both halves
  int bcol = tN * 16 + l15;      // B: N = lane&15 for both halves

  union Frag { unsigned int u[8]; v16h h; };
  v8f acc = {};
#pragma unroll
  for (int ks = 0; ks < 4; ++ks) {
    int ko = ks * 32;
    Frag a, bm;
#pragma unroll
    for (int j = 0; j < 4; ++j) {
      a.u[j]     = ru[arow * 64 + ((ko + hf * 8 + 2 * j) >> 1)];        // K pairs 0..15 blk
      a.u[4 + j] = ru[arow * 64 + ((ko + 16 + hf * 8 + 2 * j) >> 1)];   // K pairs 16..31 blk
    }
#pragma unroll
    for (int j = 0; j < 8; ++j)
      bm.u[j] = wu[bcol * 64 + ((ko + (j >> 2) * 16 + hf * 8 + 2 * (j & 3)) >> 1)];          // B K pairs
    acc = __builtin_amdgcn_wmma_f32_16x16x32_f16(false, a.h, false, bm.h,
                                                 (short)0, acc, false, false);
  }

  // D layout: VGPR g -> M = g + hf*8, N = lane&15
#pragma unroll
  for (int g = 0; g < 8; ++g) {
    int row = tM * 16 + g + hf * 8;
    if (row < M_ && bcol < M_) {
      float v = fminf(fmaxf(rintf(acc[g]), 0.0f), 1.0f);  // rint = round-half-even (jnp.round)
      outm[row * M_ + bcol] = v;
    }
  }
}

// ---------------------------------------------------------------------------
extern "C" void kernel_launch(void* const* d_in, const int* in_sizes, int n_in,
                              void* d_out, int out_size, void* d_ws, size_t ws_size,
                              hipStream_t stream) {
  (void)in_sizes; (void)n_in; (void)out_size; (void)ws_size;
  const float* y_true = (const float*)d_in[0];
  const float* bbox   = (const float*)d_in[1];
  const float* props  = (const float*)d_in[2];
  const float* mask   = (const float*)d_in[3];
  float* out = (float*)d_out;

  char* ws = (char*)d_ws;
  float* mx      = (float*)(ws);                     // 8*1000*4  = 32000 B
  int*   arg     = (int*)  (ws + 32000);             // 32000 B
  int*   slot_i  = (int*)  (ws + 64000);             // 16384 B
  int*   slot_ti = (int*)  (ws + 64000 + 16384);     // 16384 B
  int*   slot_fl = (int*)  (ws + 64000 + 32768);     // 16384 B

  k_iou   <<<dim3((N_ + 255) / 256, B_), 256, 0, stream>>>(props, bbox, mx, arg);
  k_sample<<<B_, 256, 0, stream>>>(bbox, mx, arg, slot_i, slot_ti, slot_fl);
  k_scalar<<<B_ * S_, 128, 0, stream>>>(y_true, bbox, props, slot_i, slot_ti, slot_fl, out);
  k_mask  <<<B_ * S_, 128, 0, stream>>>(mask, bbox, props, slot_i, slot_ti, slot_fl, out);
}
